// Correct_PrototypeManager_72533407695472
// MI455X (gfx1250) — matrix operand order repaired and checked by reading, not compiled, not run on hardware
//
#include <hip/hip_runtime.h>

typedef __attribute__((ext_vector_type(16))) _Float16 v16h;
typedef __attribute__((ext_vector_type(8)))  float    v8f;

#define NB   8      // batch
#define NC   256    // channels
#define NK   21     // classes
#define KP   32     // padded classes
#define HS   64     // feats spatial
#define HO   256    // output spatial

// ---------------- kernel 0: zero workspace ----------------
__global__ void zero_ws_kernel(float* __restrict__ ws, int n) {
  int i = blockIdx.x * blockDim.x + threadIdx.x;
  if (i < n) ws[i] = 0.0f;
}

// ---------------- kernel 1: per-row WMMA accumulation ----------------
// grid = (256 rows, 8 images), block = 256 threads (8 wave32)
__global__ __launch_bounds__(256)
void proto_accum_kernel(const float* __restrict__ feats,   // [8][256][64][64]
                        const float* __restrict__ preds,   // [8][21][256][256]
                        const int*   __restrict__ masks,   // [8][256][256]
                        float* __restrict__ sums,          // [8][32][256]
                        float* __restrict__ counts)        // [8][32]
{
  // tmp[c][i] holds vertical-lerp value at source x = clamp(i-1, 0, 63)
  __shared__ _Float16 tmp[NC * 66];
  __shared__ unsigned long long mcq[HO / 8];   // 256 matchClass bytes
  __shared__ unsigned int cnt_l[KP];

  const int tid = threadIdx.x;
  const int y   = blockIdx.x;    // output row
  const int b   = blockIdx.y;    // image

  if (tid < KP) cnt_l[tid] = 0u;
  __syncthreads();

  // ---- phase A: argmax over classes, match vs mask ----
  {
    const int x = tid;
    const float* p = preds + (((size_t)b * NK) * HO + y) * HO + x;
    float best = p[0];
    int bi = 0;
    #pragma unroll
    for (int k = 1; k < NK; ++k) {
      float v = p[(size_t)k * HO * HO];
      if (v > best) { best = v; bi = k; }   // strict > keeps first index (jnp.argmax)
    }
    int m = masks[((size_t)b * HO + y) * HO + x];
    // m==21 (ignore) can never equal bi<=20, so it naturally drops out
    unsigned char mc = (m == bi) ? (unsigned char)bi : (unsigned char)255;
    ((unsigned char*)mcq)[x] = mc;
    if (mc != 255u) atomicAdd(&cnt_l[mc], 1u);
  }

  // ---- phase B: vertical bilinear into LDS (half-pixel, scale 4, edge clamp) ----
  {
    const int y0 = (y - 2) >> 2;                       // floor((y+0.5)/4 - 0.5)
    const float wyt[4] = {0.625f, 0.875f, 0.125f, 0.375f};
    const float wy = wyt[y & 3];
    const int y0c = max(y0, 0);
    const int y1c = min(y0 + 1, HS - 1);
    const float* f0 = feats + (((size_t)b * NC) * HS + (size_t)y0c) * HS;
    const float* f1 = feats + (((size_t)b * NC) * HS + (size_t)y1c) * HS;
    for (int ii = tid; ii < NC * 66; ii += 256) {
      int c  = ii / 66;
      int i  = ii - c * 66;
      int xs = min(max(i - 1, 0), HS - 1);
      float a = f0[(size_t)c * (HS * HS) + xs];
      float d = f1[(size_t)c * (HS * HS) + xs];
      tmp[c * 66 + i] = (_Float16)(a + wy * (d - a));
    }
  }
  __syncthreads();

  // flush per-class counts (as float, matching f32 joint.sum in reference)
  if (tid < KP) {
    unsigned int cv = cnt_l[tid];
    if (cv) atomicAdd(&counts[b * KP + tid], (float)cv);
  }

  // ---- phase C: WMMA  sums[k,c] += onehot[k,px] * feats_up[c,px] ----
  const int lane  = tid & 31;
  const int w     = tid >> 5;       // wave id 0..7 -> channel tiles 2w, 2w+1
  const int nIdx  = lane & 15;      // A: row M; B: column N; D: column N
  const int hs    = lane >> 4;      // half-select
  const int cbase = w * 32;

  v8f acc0 = {}; v8f acc1 = {}; v8f acc2 = {}; v8f acc3 = {};
  const _Float16 W4[4] = {(_Float16)0.625f, (_Float16)0.875f,
                          (_Float16)0.125f, (_Float16)0.375f};

  for (int g = 0; g < 8; ++g) {     // 8 groups of 32 pixels
    const int gx = g * 32;

    // --- A fragments (16x32 f16 layout): lane m=nIdx; hs=0 -> K {0..15}, hs=1 -> K {16..31}
    const unsigned long long qa = mcq[g * 4 + 2 * hs];       // bytes K = hs*16 + 0..7
    const unsigned long long qb = mcq[g * 4 + 2 * hs + 1];   // bytes K = hs*16 + 8 + 0..7
    v16h a0, a1;
    #pragma unroll
    for (int e = 0; e < 8; ++e) {
      unsigned int by1 = (unsigned int)(qa >> (8 * e)) & 0xffu;
      unsigned int by2 = (unsigned int)(qb >> (8 * e)) & 0xffu;
      a0[e]     = (by1 == (unsigned)nIdx)        ? (_Float16)1.0f : (_Float16)0.0f;
      a1[e]     = (by1 == (unsigned)(nIdx + 16)) ? (_Float16)1.0f : (_Float16)0.0f;
      a0[8 + e] = (by2 == (unsigned)nIdx)        ? (_Float16)1.0f : (_Float16)0.0f;
      a1[8 + e] = (by2 == (unsigned)(nIdx + 16)) ? (_Float16)1.0f : (_Float16)0.0f;
    }

    // --- B fragments (32x16 f16 layout): lane n=nIdx, VGPR v halves = K 2v,2v+1 (+16 for hs=1)
    v16h bf[2];
    #pragma unroll
    for (int t2 = 0; t2 < 2; ++t2) {
      const int cn = cbase + t2 * 16 + nIdx;
      const int X0 = gx + hs * 16;
      const int Q  = (X0 + 2) >> 2;
      const _Float16* tr = tmp + cn * 66 + Q;
      _Float16 h[6];
      #pragma unroll
      for (int t = 0; t < 6; ++t) h[t] = tr[t];
      v16h bb;
      #pragma unroll
      for (int e = 0; e < 16; ++e) {
        int mI = (e + 2) >> 2;                 // tap pair selector 0..4
        _Float16 lo = h[mI], hi = h[mI + 1];
        bb[e] = (_Float16)(lo + W4[e & 3] * (hi - lo));
      }
      bf[t2] = bb;
    }

    acc0 = __builtin_amdgcn_wmma_f32_16x16x32_f16(false, a0, false, bf[0], (short)0, acc0, false, false);
    acc1 = __builtin_amdgcn_wmma_f32_16x16x32_f16(false, a1, false, bf[0], (short)0, acc1, false, false);
    acc2 = __builtin_amdgcn_wmma_f32_16x16x32_f16(false, a0, false, bf[1], (short)0, acc2, false, false);
    acc3 = __builtin_amdgcn_wmma_f32_16x16x32_f16(false, a1, false, bf[1], (short)0, acc3, false, false);
  }

  // ---- write-out: D layout VGPR i -> M = i + 8*hs, N = nIdx ----
  float* sb = sums + (size_t)b * KP * NC;
  #pragma unroll
  for (int i = 0; i < 8; ++i) {
    const int Mrow = i + hs * 8;
    atomicAdd(&sb[(size_t)(Mrow)      * NC + cbase + nIdx],      acc0[i]);
    atomicAdd(&sb[(size_t)(16 + Mrow) * NC + cbase + nIdx],      acc1[i]);
    atomicAdd(&sb[(size_t)(Mrow)      * NC + cbase + 16 + nIdx], acc2[i]);
    atomicAdd(&sb[(size_t)(16 + Mrow) * NC + cbase + 16 + nIdx], acc3[i]);
  }
}

// ---------------- kernel 2: normalize + batch mean ----------------
__global__ void finalize_kernel(const float* __restrict__ sums,
                                const float* __restrict__ counts,
                                float* __restrict__ out)
{
  const int k = blockIdx.x;     // 0..20
  const int c = threadIdx.x;    // 0..255
  float acc = 0.0f;
  #pragma unroll
  for (int b = 0; b < NB; ++b)
    acc += sums[((size_t)b * KP + k) * NC + c] / (counts[b * KP + k] + 1e-6f);
  out[k * NC + c] = acc * 0.125f;
}

extern "C" void kernel_launch(void* const* d_in, const int* in_sizes, int n_in,
                              void* d_out, int out_size, void* d_ws, size_t ws_size,
                              hipStream_t stream) {
  const float* feats = (const float*)d_in[0];
  const float* preds = (const float*)d_in[1];
  const int*   masks = (const int*)d_in[2];
  float* sums   = (float*)d_ws;                 // 8*32*256 floats
  float* counts = sums + NB * KP * NC;          // 8*32 floats
  float* out    = (float*)d_out;                // [21][256]

  const int nz = NB * KP * NC + NB * KP;
  zero_ws_kernel<<<(nz + 255) / 256, 256, 0, stream>>>(sums, nz);
  proto_accum_kernel<<<dim3(HO, NB), 256, 0, stream>>>(feats, preds, masks, sums, counts);
  finalize_kernel<<<NK, 256, 0, stream>>>(sums, counts, out);
}
